// model_27152783245996
// MI455X (gfx1250) — matrix operand order repaired and checked by reading, not compiled, not run on hardware
//
#include <hip/hip_runtime.h>
#include <hip/hip_bf16.h>

// ---------------------------------------------------------------------------
// Types for CDNA5 WMMA (wave32): v_wmma_f32_16x16x32_bf16
// ---------------------------------------------------------------------------
typedef __bf16 bf16;
typedef __attribute__((ext_vector_type(16))) __bf16 v16bf;
typedef __attribute__((ext_vector_type(8)))  float  v8f;

static __device__ __forceinline__ v8f v8zero() {
    v8f z;
#pragma unroll
    for (int i = 0; i < 8; ++i) z[i] = 0.f;
    return z;
}

// A fragment: 16x32 (MxK) bf16, row-major source with leading dim `ld` (elems).
// ISA layout: frag[0..7] = row[h*8 .. h*8+7], frag[8..15] = row[16+h*8 .. +7]
static __device__ __forceinline__ v16bf load_frag_a(const bf16* base, int ld, int lane) {
    int r = lane & 15, h = lane >> 4;
    const bf16* rowp = base + r * ld;
    union { uint4 u[2]; v16bf v; } t;
    t.u[0] = *(const uint4*)(rowp + h * 8);
    t.u[1] = *(const uint4*)(rowp + 16 + h * 8);
    return t.v;
}

// B fragment: 32x16 (KxN) bf16 stored N-major: base[n*ld + k] (ld >= 32).
// frag[0..7] = K[h*8 .. h*8+7], frag[8..15] = K[16+h*8 .. +7], column N = lane&15.
static __device__ __forceinline__ v16bf load_frag_b(const bf16* base, int ld, int lane) {
    int n = lane & 15, h = lane >> 4;
    const bf16* rowp = base + n * ld;
    union { uint4 u[2]; v16bf v; } t;
    t.u[0] = *(const uint4*)(rowp + h * 8);
    t.u[1] = *(const uint4*)(rowp + 16 + h * 8);
    return t.v;
}

static __device__ __forceinline__ v8f wmma_bf16(v16bf a, v16bf b, v8f c) {
    return __builtin_amdgcn_wmma_f32_16x16x32_bf16(false, a, false, b, (short)0, c, false, false);
}

// Async global -> LDS copy (CDNA5 async engine, tracked by ASYNCcnt).
// lds_off: byte offset within the wave's LDS allocation (flat-ptr low 32 bits).
static __device__ __forceinline__ void async_b128(unsigned lds_off, const void* gaddr) {
    asm volatile("global_load_async_to_lds_b128 %0, %1, off"
                 :: "v"(lds_off), "v"(gaddr) : "memory");
}

// ---------------------------------------------------------------------------
// Constants
// ---------------------------------------------------------------------------
#define PADW    258          // padded width/height
#define PAD_E   (258*258*64) // elems per padded NHWC bf16 buffer
#define SP      2048
#define NEDGE   12288

// ---------------------------------------------------------------------------
// Utility kernels
// ---------------------------------------------------------------------------
__global__ void fill_zero_u32(unsigned int* p, int n) {
    int g = blockIdx.x * blockDim.x + threadIdx.x;
    if (g < n) p[g] = 0u;
}

__global__ void f32_to_bf16_k(const float* in, bf16* out, int n) {
    int g = blockIdx.x * blockDim.x + threadIdx.x;
    if (g < n) out[g] = (bf16)in[g];
}

// OIHW fp32 (64,64,3,3) -> bf16 [tap(9)][chunk(2)][oc(64)][ic(32)]
__global__ void repack_conv_k(const float* w, bf16* dst) {
    int g = blockIdx.x * blockDim.x + threadIdx.x;
    if (g >= 9 * 2 * 64 * 32) return;
    int ic = g & 31;
    int o  = (g >> 5) & 63;
    int tc = g >> 11;            // tap*2 + chunk
    int t = tc >> 1, c = tc & 1;
    int ty = t / 3, tx = t % 3;
    dst[g] = (bf16)w[((o * 64 + (c * 32 + ic)) * 3 + ty) * 3 + tx];
}

// (N,K) fp32 row-major -> (Npad,K) bf16, zero rows beyond N
__global__ void repack_gcnw_k(const float* w, bf16* dst, int N, int K, int Npad) {
    int g = blockIdx.x * blockDim.x + threadIdx.x;
    if (g >= Npad * K) return;
    int n = g / K;
    dst[g] = (n < N) ? (bf16)w[g] : (bf16)0.f;
}

// ---------------------------------------------------------------------------
// Stem: 7x7 s2 conv 3->64 + BN + ReLU, 1024^2 -> 512^2, NHWC bf16 out
// ---------------------------------------------------------------------------
__global__ __launch_bounds__(256) void stem_conv7(
    const float* __restrict__ img, const float* __restrict__ w,
    const float* __restrict__ gam, const float* __restrict__ bet,
    const float* __restrict__ mea, const float* __restrict__ var_,
    bf16* __restrict__ out)
{
    int g = blockIdx.x * 256 + threadIdx.x;        // 512*512*64 threads
    int oc = g & 63;
    int pix = g >> 6;
    int x = pix & 511, y = pix >> 9;
    float a = 0.f;
#pragma unroll
    for (int c = 0; c < 3; ++c) {
        const float* ip = img + c * 1024 * 1024;
        const float* wp = w + (oc * 3 + c) * 49;
        for (int ky = 0; ky < 7; ++ky) {
            int iy = y * 2 - 3 + ky;
            if (iy < 0 || iy > 1023) continue;
            for (int kx = 0; kx < 7; ++kx) {
                int ix = x * 2 - 3 + kx;
                if (ix < 0 || ix > 1023) continue;
                a = fmaf(ip[iy * 1024 + ix], wp[ky * 7 + kx], a);
            }
        }
    }
    float sc = gam[oc] * rsqrtf(var_[oc] + 1e-5f);
    out[g] = (bf16)fmaxf(a * sc + (bet[oc] - mea[oc] * sc), 0.f);
}

// maxpool 3x3 s2 pad1: 512^2 -> 256^2, write into padded NHWC bf16 interior
__global__ __launch_bounds__(256) void maxpool_k(const bf16* __restrict__ in,
                                                 bf16* __restrict__ outPad)
{
    int g = blockIdx.x * 256 + threadIdx.x;        // 256*256*64 threads
    int oc = g & 63;
    int pix = g >> 6;
    int x = pix & 255, y = pix >> 8;
    float mx = -3.4e38f;
    for (int ky = 0; ky < 3; ++ky) {
        int sy = 2 * y - 1 + ky;
        if (sy < 0 || sy > 511) continue;
        for (int kx = 0; kx < 3; ++kx) {
            int sx = 2 * x - 1 + kx;
            if (sx < 0 || sx > 511) continue;
            mx = fmaxf(mx, (float)in[(sy * 512 + sx) * 64 + oc]);
        }
    }
    outPad[((y + 1) * PADW + (x + 1)) * 64 + oc] = (bf16)mx;
}

// ---------------------------------------------------------------------------
// 3x3 conv 64->64 implicit-GEMM via WMMA bf16, fused BN (+res) + ReLU.
// 256 threads = 8 waves; tile 128 pixels x 64 oc; wave owns 32x32.
// K loop = 9 taps x 2 chunks of 32 ic. Triple-buffered LDS fed by the CDNA5
// async global->LDS engine; ONE barrier per K-step:
//   s_wait_asynccnt<=3 (stage s landed, stage s+1 in flight) -> barrier ->
//   issue stage s+2 -> WMMA on stage s.
// ---------------------------------------------------------------------------
#define STG_E  6144   // elems per LDS stage buffer: A 128*32 + B 64*32
#define STG_B  12288  // bytes per stage buffer

__global__ __launch_bounds__(256) void conv3x3_wmma(
    const bf16* __restrict__ in,   // padded NHWC bf16 (258x258x64)
    const bf16* __restrict__ wk,   // [9][2][64][32] bf16
    const float* __restrict__ gam, const float* __restrict__ bet,
    const float* __restrict__ mea, const float* __restrict__ var_,
    const bf16* __restrict__ res,  // padded residual (may be null)
    bf16* __restrict__ out)        // padded NHWC bf16
{
    __shared__ __align__(16) bf16 lds[3 * STG_E];   // 36 KB, triple buffered

    int tid  = threadIdx.x;
    int lane = tid & 31, wave = tid >> 5;
    int wm = wave & 3, wn = wave >> 2;
    int wg = blockIdx.x;                 // 512 workgroups
    int y = wg >> 1, x0 = (wg & 1) * 128;

    // LDS byte offset of buffer base (flat shared ptr low 32 bits == LDS addr)
    unsigned ldsbase = (unsigned)(unsigned long long)(&lds[0]);

    // per-thread stage issue: 2x b128 for A tile (32 B), 1x b128 for B tile
    auto issue = [&](int s) {
        int b = s % 3;
        int t = s >> 1, c = s & 1;
        int dy = t / 3, dx = t % 3;
        int p = tid >> 1, hc = tid & 1;
        unsigned dA = ldsbase + (unsigned)(b * STG_B) + (unsigned)(p * 64 + hc * 32);
        const bf16* sA = in + ((y + dy) * PADW + (x0 + dx + p)) * 64 + c * 32 + hc * 16;
        async_b128(dA,      sA);
        async_b128(dA + 16, sA + 8);
        unsigned dB = ldsbase + (unsigned)(b * STG_B) + 8192u + (unsigned)(tid * 16);
        const bf16* sB = wk + (t * 2 + c) * 2048 + tid * 8;
        async_b128(dB, sB);
    };

    v8f acc[2][2];
    acc[0][0] = v8zero(); acc[0][1] = v8zero();
    acc[1][0] = v8zero(); acc[1][1] = v8zero();

    issue(0);
    issue(1);
    for (int s = 0; s < 18; ++s) {
        if (s < 17) asm volatile("s_wait_asynccnt 0x3" ::: "memory");
        else        asm volatile("s_wait_asynccnt 0x0" ::: "memory");
        __syncthreads();                  // stage s visible to all waves
        if (s + 2 < 18) issue(s + 2);     // prefetch into third buffer

        const bf16* ldsA = &lds[(s % 3) * STG_E];
        const bf16* ldsB = &lds[(s % 3) * STG_E + 4096];
        v16bf a0 = load_frag_a(ldsA + (wm * 32) * 32,      32, lane);
        v16bf a1 = load_frag_a(ldsA + (wm * 32 + 16) * 32, 32, lane);
        v16bf b0 = load_frag_b(ldsB + (wn * 32) * 32,      32, lane);
        v16bf b1 = load_frag_b(ldsB + (wn * 32 + 16) * 32, 32, lane);
        acc[0][0] = wmma_bf16(a0, b0, acc[0][0]);
        acc[0][1] = wmma_bf16(a0, b1, acc[0][1]);
        acc[1][0] = wmma_bf16(a1, b0, acc[1][0]);
        acc[1][1] = wmma_bf16(a1, b1, acc[1][1]);
    }

    // epilogue: BN (+res) + ReLU, bf16 NHWC into padded interior
    int hh = lane >> 4, nn = lane & 15;
#pragma unroll
    for (int ni = 0; ni < 2; ++ni) {
        int oc = wn * 32 + ni * 16 + nn;
        float sc = gam[oc] * rsqrtf(var_[oc] + 1e-5f);
        float bi = bet[oc] - mea[oc] * sc;
#pragma unroll
        for (int mi = 0; mi < 2; ++mi) {
#pragma unroll
            for (int j = 0; j < 8; ++j) {
                int p = wm * 32 + mi * 16 + j + 8 * hh;
                int idx = ((y + 1) * PADW + (x0 + p + 1)) * 64 + oc;
                float val = acc[mi][ni][j] * sc + bi;
                if (res) val += (float)res[idx];
                out[idx] = (bf16)fmaxf(val, 0.f);
            }
        }
    }
}

// ---------------------------------------------------------------------------
// Fused bilinear upsample (align_corners, 256->1024) + superpixel mean scatter.
// One thread = (output pixel, channel pair). node must be pre-zeroed.
// ---------------------------------------------------------------------------
__global__ __launch_bounds__(256) void upsample_segment_k(
    const bf16* __restrict__ fea,     // padded NHWC bf16 (final features)
    const int* __restrict__ labels,
    float* __restrict__ node)         // [SP][64] f32 accumulators
{
    int g = blockIdx.x * 256 + threadIdx.x;   // 1024*1024*32
    int lane = g & 31;
    int pix  = g >> 5;
    int ox = pix & 1023, oy = pix >> 10;
    const float s = 255.0f / 1023.0f;
    float fy = oy * s, fx = ox * s;
    int y0 = (int)fy, x0 = (int)fx;
    float wy = fy - (float)y0, wx = fx - (float)x0;
    int y1 = y0 + 1; if (y1 > 255) y1 = 255;
    int x1 = x0 + 1; if (x1 > 255) x1 = 255;
    int ch = lane * 2;

    const bf16* p00 = fea + ((y0 + 1) * PADW + x0 + 1) * 64 + ch;
    const bf16* p01 = fea + ((y0 + 1) * PADW + x1 + 1) * 64 + ch;
    const bf16* p10 = fea + ((y1 + 1) * PADW + x0 + 1) * 64 + ch;
    const bf16* p11 = fea + ((y1 + 1) * PADW + x1 + 1) * 64 + ch;
    float w00 = (1.f - wy) * (1.f - wx), w01 = (1.f - wy) * wx;
    float w10 = wy * (1.f - wx),         w11 = wy * wx;
    float v0 = w00 * (float)p00[0] + w01 * (float)p01[0] + w10 * (float)p10[0] + w11 * (float)p11[0];
    float v1 = w00 * (float)p00[1] + w01 * (float)p01[1] + w10 * (float)p10[1] + w11 * (float)p11[1];

    const float inv = 1.0f / (1024.0f * 1024.0f);
    int lab = labels[pix];
    atomicAdd(&node[lab * 64 + ch],     v0 * inv);
    atomicAdd(&node[lab * 64 + ch + 1], v1 * inv);
}

// GCN edge scatter: agg[dst] += h[src]   (agg pre-zeroed)
__global__ void gcn_scatter_k(const float* __restrict__ h, const int* __restrict__ edges,
                              int cshift, float* __restrict__ agg)
{
    int g = blockIdx.x * blockDim.x + threadIdx.x;
    int C = 1 << cshift;
    if (g >= NEDGE * C) return;
    int e = g >> cshift, ch = g & (C - 1);
    int s = edges[e], d = edges[NEDGE + e];
    atomicAdd(&agg[d * C + ch], h[s * C + ch]);
}

// ---------------------------------------------------------------------------
// Small GEMM for GCN linears via WMMA: out = relu(A(MxK) * W^T + bias)
// W stored (Npad x K) bf16 row-major. One wave per 16x16 tile, 8 waves/block.
// ---------------------------------------------------------------------------
__global__ __launch_bounds__(256) void gemm_wmma_k(
    const bf16* __restrict__ A, const bf16* __restrict__ W,
    const float* __restrict__ bias, float* __restrict__ out,
    int M, int K, int Npad, int Nout)
{
    int lane = threadIdx.x & 31;
    int wave = threadIdx.x >> 5;
    int nt = Npad >> 4;
    int wt = blockIdx.x * 8 + wave;
    if (wt >= (M >> 4) * nt) return;
    int m0 = (wt / nt) << 4;
    int n0 = (wt % nt) << 4;

    v8f acc = v8zero();
    for (int k0 = 0; k0 < K; k0 += 32) {
        v16bf a = load_frag_a(A + m0 * K + k0, K, lane);
        v16bf b = load_frag_b(W + n0 * K + k0, K, lane);
        acc = wmma_bf16(a, b, acc);
    }
    int hh = lane >> 4, nn = lane & 15;
    int oc = n0 + nn;
    if (oc < Nout) {
        float bi = bias[oc];
#pragma unroll
        for (int j = 0; j < 8; ++j)
            out[(m0 + j + 8 * hh) * Nout + oc] = fmaxf(acc[j] + bi, 0.f);
    }
}

// ---------------------------------------------------------------------------
// Host launcher
// ---------------------------------------------------------------------------
extern "C" void kernel_launch(void* const* d_in, const int* in_sizes, int n_in,
                              void* d_out, int out_size, void* d_ws, size_t ws_size,
                              hipStream_t stream)
{
    // ---- resolve flattened input pytree (sorted-key jax order vs insertion order)
    const float *image, *stemw, *sg, *sb, *sm, *sv;
    const int *labels, *edges;
    const float *bc1[3], *bc2[3];
    const float *b1g[3], *b1b[3], *b1m[3], *b1v[3];
    const float *b2g[3], *b2b[3], *b2m[3], *b2v[3];
    const float *g1w, *g1b, *g2w, *g2b, *g3w, *g3b;
    auto F = [&](int i) { return (const float*)d_in[i]; };

    if (n_in > 0 && in_sizes[0] == 2 * NEDGE) {
        // sorted dict-key pytree order
        edges = (const int*)d_in[0]; image = F(1); labels = (const int*)d_in[2];
        for (int i = 0; i < 3; ++i) {
            int b = 3 + i * 10;
            b1b[i] = F(b);     b1g[i] = F(b + 1); b1m[i] = F(b + 2); b1v[i] = F(b + 3);
            b2b[i] = F(b + 4); b2g[i] = F(b + 5); b2m[i] = F(b + 6); b2v[i] = F(b + 7);
            bc1[i] = F(b + 8); bc2[i] = F(b + 9);
        }
        sb = F(33); sg = F(34); sm = F(35); sv = F(36); stemw = F(37);
        g1b = F(38); g1w = F(39); g2b = F(40); g2w = F(41); g3b = F(42); g3w = F(43);
    } else {
        // insertion order
        image = F(0); stemw = F(1);
        sg = F(2); sb = F(3); sm = F(4); sv = F(5);
        for (int i = 0; i < 3; ++i) {
            int b = 6 + i * 10;
            bc1[i] = F(b);
            b1g[i] = F(b + 1); b1b[i] = F(b + 2); b1m[i] = F(b + 3); b1v[i] = F(b + 4);
            bc2[i] = F(b + 5);
            b2g[i] = F(b + 6); b2b[i] = F(b + 7); b2m[i] = F(b + 8); b2v[i] = F(b + 9);
        }
        g1w = F(36); g1b = F(37); g2w = F(38); g2b = F(39); g3w = F(40); g3b = F(41);
        labels = (const int*)d_in[42]; edges = (const int*)d_in[43];
    }

    // ---- workspace carve
    char* ws = (char*)d_ws;
    size_t off = 0;
    auto alloc = [&](size_t bytes) { char* p = ws + off; off = (off + bytes + 255) & ~(size_t)255; return p; };
    bf16* padA = (bf16*)alloc(PAD_E * 2);
    bf16* padB = (bf16*)alloc(PAD_E * 2);
    bf16* padC = (bf16*)alloc(PAD_E * 2);
    bf16* stem = (bf16*)alloc((size_t)512 * 512 * 64 * 2);
    bf16* wblk[6];
    for (int i = 0; i < 6; ++i) wblk[i] = (bf16*)alloc(9 * 2 * 64 * 32 * 2);
    float* node = (float*)alloc(SP * 64 * 4);
    float* agg  = (float*)alloc(SP * 64 * 4);
    float* h1   = (float*)alloc(SP * 64 * 4);
    float* h2   = (float*)alloc(SP * 32 * 4);
    bf16*  hbf  = (bf16*)alloc(SP * 64 * 2);
    bf16*  wg1  = (bf16*)alloc(64 * 64 * 2);
    bf16*  wg2  = (bf16*)alloc(32 * 64 * 2);
    bf16*  wg3  = (bf16*)alloc(32 * 32 * 2);
    (void)ws_size;

    auto nb = [](long n, int b) { return (int)((n + b - 1) / b); };

    // ---- zero padded buffers + node accumulator
    fill_zero_u32<<<nb(PAD_E / 2, 256), 256, 0, stream>>>((unsigned*)padA, PAD_E / 2);
    fill_zero_u32<<<nb(PAD_E / 2, 256), 256, 0, stream>>>((unsigned*)padB, PAD_E / 2);
    fill_zero_u32<<<nb(PAD_E / 2, 256), 256, 0, stream>>>((unsigned*)padC, PAD_E / 2);
    fill_zero_u32<<<nb(SP * 64, 256), 256, 0, stream>>>((unsigned*)node, SP * 64);

    // ---- weight repack
    for (int i = 0; i < 3; ++i) {
        repack_conv_k<<<nb(36864, 256), 256, 0, stream>>>(bc1[i], wblk[2 * i]);
        repack_conv_k<<<nb(36864, 256), 256, 0, stream>>>(bc2[i], wblk[2 * i + 1]);
    }
    repack_gcnw_k<<<nb(64 * 64, 256), 256, 0, stream>>>(g1w, wg1, 64, 64, 64);
    repack_gcnw_k<<<nb(32 * 64, 256), 256, 0, stream>>>(g2w, wg2, 32, 64, 32);
    repack_gcnw_k<<<nb(32 * 32, 256), 256, 0, stream>>>(g3w, wg3, 21, 32, 32);

    // ---- stem + maxpool
    stem_conv7<<<512 * 512 * 64 / 256, 256, 0, stream>>>(image, stemw, sg, sb, sm, sv, stem);
    maxpool_k<<<256 * 256 * 64 / 256, 256, 0, stream>>>(stem, padA);

    // ---- layer1: 3 BasicBlocks (A->B->C, C->B->A, A->B->C), all WMMA
    conv3x3_wmma<<<512, 256, 0, stream>>>(padA, wblk[0], b1g[0], b1b[0], b1m[0], b1v[0], nullptr, padB);
    conv3x3_wmma<<<512, 256, 0, stream>>>(padB, wblk[1], b2g[0], b2b[0], b2m[0], b2v[0], padA,    padC);
    conv3x3_wmma<<<512, 256, 0, stream>>>(padC, wblk[2], b1g[1], b1b[1], b1m[1], b1v[1], nullptr, padB);
    conv3x3_wmma<<<512, 256, 0, stream>>>(padB, wblk[3], b2g[1], b2b[1], b2m[1], b2v[1], padC,    padA);
    conv3x3_wmma<<<512, 256, 0, stream>>>(padA, wblk[4], b1g[2], b1b[2], b1m[2], b1v[2], nullptr, padB);
    conv3x3_wmma<<<512, 256, 0, stream>>>(padB, wblk[5], b2g[2], b2b[2], b2m[2], b2v[2], padA,    padC);

    // ---- upsample + superpixel mean
    upsample_segment_k<<<1024 * 1024 * 32 / 256, 256, 0, stream>>>(padC, labels, node);

    // ---- GCN layer 1: 64 -> 64
    fill_zero_u32<<<nb(SP * 64, 256), 256, 0, stream>>>((unsigned*)agg, SP * 64);
    gcn_scatter_k<<<nb((long)NEDGE * 64, 256), 256, 0, stream>>>(node, edges, 6, agg);
    f32_to_bf16_k<<<nb(SP * 64, 256), 256, 0, stream>>>(agg, hbf, SP * 64);
    gemm_wmma_k<<<nb((SP / 16) * 4, 8), 256, 0, stream>>>(hbf, wg1, g1b, h1, SP, 64, 64, 64);

    // ---- GCN layer 2: 64 -> 32
    fill_zero_u32<<<nb(SP * 64, 256), 256, 0, stream>>>((unsigned*)agg, SP * 64);
    gcn_scatter_k<<<nb((long)NEDGE * 64, 256), 256, 0, stream>>>(h1, edges, 6, agg);
    f32_to_bf16_k<<<nb(SP * 64, 256), 256, 0, stream>>>(agg, hbf, SP * 64);
    gemm_wmma_k<<<nb((SP / 16) * 2, 8), 256, 0, stream>>>(hbf, wg2, g2b, h2, SP, 64, 32, 32);

    // ---- GCN layer 3: 32 -> 21 (N padded to 32, guarded store to d_out)
    fill_zero_u32<<<nb(SP * 32, 256), 256, 0, stream>>>((unsigned*)agg, SP * 32);
    gcn_scatter_k<<<nb((long)NEDGE * 32, 256), 256, 0, stream>>>(h2, edges, 5, agg);
    f32_to_bf16_k<<<nb(SP * 32, 256), 256, 0, stream>>>(agg, hbf, SP * 32);
    gemm_wmma_k<<<nb((SP / 16) * 2, 8), 256, 0, stream>>>(hbf, wg3, g3b, (float*)d_out, SP, 32, 32, 21);
    (void)out_size; (void)n_in;
}
